// CorrespondenceGeneration_10101763080271
// MI455X (gfx1250) — matrix operand order repaired and checked by reading, not compiled, not run on hardware
//
#include <hip/hip_runtime.h>
#include <hip/hip_bf16.h>

typedef __attribute__((ext_vector_type(16))) _Float16 v16h;
typedef __attribute__((ext_vector_type(8)))  _Float16 v8h;
typedef __attribute__((ext_vector_type(8)))  float    v8f;

#define NB      2
#define CCH     64
#define HH      96
#define WW      96
#define HW      (HH*WW)
#define PS      3
#define HP      94
#define WP      94
#define NPATCH  (HP*WP)          // 8836
#define KDIM    (CCH*PS*PS)      // 576 = 18*32
#define KCH     (KDIM/32)        // 18
#define NPAD    8960             // 35*256 (M padding), also multiple of 64
#define BM      256              // rows per block (16 waves x 16)
#define MBLK    (NPAD/BM)        // 35
#define RTILES  (NPAD/64)        // 140 r-tiles of 64
#define RSPLIT  4
#define TPC     (RTILES/RSPLIT)  // 35 tiles per block (exact)

// async B staging pipeline
#define CHK     6                     // k-steps (x32 halves) per chunk
#define NCHUNK  (KCH/CHK)             // 3 chunks per r-tile
#define ROWB    80                    // padded bytes per 64-byte row (bank-conflict-free)
#define STEPB   (64*ROWB)             // 5120 B per k-step slab
#define CHUNKB  (CHK*STEPB)           // 30720 B per chunk
#define NBUF    3
#define SMEMB   (NBUF*CHUNKB)         // 92160 B dynamic LDS

// ---------------------------------------------------------------- utilities
__global__ void zero_best_kernel(unsigned long long* __restrict__ best, int count) {
    int t = blockIdx.x * blockDim.x + threadIdx.x;
    if (t < count) best[t] = 0ull;
}

__global__ void normalize_kernel(const float* __restrict__ in, float* __restrict__ out) {
    int t = blockIdx.x * blockDim.x + threadIdx.x;
    if (t >= NB * HW) return;
    int n = t / HW, q = t % HW;
    const float* ip = in  + (size_t)n * CCH * HW + q;
    float*       op = out + (size_t)n * CCH * HW + q;
    float ss = 0.f;
    for (int c = 0; c < CCH; ++c) { float v = ip[(size_t)c * HW]; ss += v * v; }
    float s = 1.0f / fmaxf(sqrtf(ss), 1e-12f);
    for (int c = 0; c < CCH; ++c) op[(size_t)c * HW] = ip[(size_t)c * HW] * s;
}

__global__ void build_patches_kernel(const float* __restrict__ normF, _Float16* __restrict__ dst) {
    int t = blockIdx.x * blockDim.x + threadIdx.x;
    if (t >= NB * NPAD * CCH) return;
    int c = t % CCH;
    int i = (t / CCH) % NPAD;
    int n = t / (CCH * NPAD);
    _Float16* drow = dst + ((size_t)n * NPAD + i) * KDIM + c * 9;
    if (i < NPATCH) {
        int y = i / WP, x = i % WP;
        const float* src = normF + ((size_t)n * CCH + c) * HW + y * WW + x;
#pragma unroll
        for (int p = 0; p < 9; ++p) {
            int dy = p / 3, dx = p % 3;
            drow[p] = (_Float16)src[dy * WW + dx];
        }
    } else {
#pragma unroll
        for (int p = 0; p < 9; ++p) drow[p] = (_Float16)0.f;
    }
}

__global__ void patch_scale_kernel(const float* __restrict__ normF2, float* __restrict__ scaleBuf) {
    int r = blockIdx.x, n = blockIdx.y, lane = threadIdx.x;
    int y = r / WP, x = r % WP;
    const float* base = normF2 + (size_t)n * CCH * HW;
    float ss = 0.f;
    for (int k = lane; k < KDIM; k += 32) {
        int c = k / 9, p = k % 9;
        int dy = p / 3, dx = p % 3;
        float v = base[(size_t)c * HW + (y + dy) * WW + (x + dx)];
        ss += v * v;
    }
#pragma unroll
    for (int m = 16; m >= 1; m >>= 1) ss += __shfl_xor(ss, m, 32);
    if (lane == 0) scaleBuf[(size_t)n * NPAD + r] = 1.0f / (sqrtf(ss) + 1e-5f);
}

// ------------------------------------------------- WMMA correlation + argmax
__device__ __forceinline__ void stage_chunk_async(unsigned long long bbase64, unsigned rbase,
                                                  unsigned kbase, unsigned ldsBuf, unsigned tid)
{
    // chunk = 6 k-steps x 64 rows x 64B; each thread moves 3 x 16B (512 thr * 3 = 1536 xfers)
#pragma unroll
    for (int rep = 0; rep < 3; ++rep) {
        unsigned j   = tid + (unsigned)rep * 512u;
        unsigned s   = j >> 8;          // k-step 0..5
        unsigned row = (j >> 2) & 63u;  // 0..63
        unsigned q   = j & 3u;          // 16B quarter of the 64B row
        unsigned lds = ldsBuf + s * (unsigned)STEPB + row * (unsigned)ROWB + q * 16u;
        unsigned gof = ((rbase + row) * (unsigned)KDIM + kbase + s * 32u) * 2u + q * 16u;
        asm volatile("global_load_async_to_lds_b128 %0, %1, %2"
                     :
                     : "v"(lds), "v"(gof), "s"(bbase64)
                     : "memory");
    }
}

__global__ __launch_bounds__(512) void corr_argmax_kernel(
    const _Float16* __restrict__ Apat, const _Float16* __restrict__ Bpat,
    const float* __restrict__ scaleBuf, unsigned long long* __restrict__ best)
{
    extern __shared__ char smem[];
    const unsigned smemBase = (unsigned)(size_t)smem;

    const int n    = blockIdx.z;
    const unsigned tid = threadIdx.x;
    const int wave = tid >> 5;
    const int lid  = tid & 31;
    const int lo16 = lid & 15;
    const int hi16 = lid >> 4;
    const unsigned Mbase = blockIdx.x * (unsigned)BM + (unsigned)wave * 16u;

    // ---- persistent A panel: 16 rows x 576 K in 18 fragments -------------
    // lane L(<16): M=L, elems 0-15 = K{b..b+15}; lanes 16-31: +16.
    const unsigned rowA = Mbase + (unsigned)lo16;
    const _Float16* arow = Apat + ((size_t)n * NPAD + rowA) * KDIM + hi16 * 16;
    v16h afrag[KCH];
#pragma unroll
    for (int k = 0; k < KCH; ++k) {
        v8h l = *(const v8h*)(arow + k * 32);
        v8h h = *(const v8h*)(arow + k * 32 + 8);
        afrag[k] = __builtin_shufflevector(l, h, 0,1,2,3,4,5,6,7,8,9,10,11,12,13,14,15);
    }

    float    bestVal[8];
    unsigned bestIdx[8];
#pragma unroll
    for (int v = 0; v < 8; ++v) { bestVal[v] = -3.4e38f; bestIdx[v] = 0xFFFFFFFFu; }

    const int t0 = blockIdx.y * TPC;
    const unsigned long long bbase64 =
        (unsigned long long)(size_t)(Bpat + (size_t)n * NPAD * KDIM);
    const float* sbase = scaleBuf + (size_t)n * NPAD;

    const int G = TPC * NCHUNK;  // 105 chunks in this block's pipeline

    // prologue: stage chunks 0 and 1
    stage_chunk_async(bbase64, (unsigned)t0 * 64u, 0u * CHK * 32u, smemBase + 0u * CHUNKB, tid);
    stage_chunk_async(bbase64, (unsigned)t0 * 64u, 1u * CHK * 32u, smemBase + 1u * CHUNKB, tid);

    int g = 0;
    for (int t = t0; t < t0 + TPC; ++t) {
        const unsigned rbase = (unsigned)t * 64u;
        v8f acc[4];
#pragma unroll
        for (int ns = 0; ns < 4; ++ns) acc[ns] = (v8f){};

        for (int c = 0; c < NCHUNK; ++c, ++g) {
            // oldest in-flight chunk (g) must have landed; newest (g+1) may still fly
            if (g == G - 1) asm volatile("s_wait_asynccnt 0x0" ::: "memory");
            else            asm volatile("s_wait_asynccnt 0x3" ::: "memory");
            __syncthreads();   // chunk g visible to all; buffer (g+2)%3 free (chunk g-1 consumed)

            if (g + 2 < G) {
                int gg = g + 2;
                unsigned rb = (unsigned)(t0 + gg / NCHUNK) * 64u;
                unsigned kb = (unsigned)(gg % NCHUNK) * CHK * 32u;
                stage_chunk_async(bbase64, rb, kb, smemBase + (unsigned)(gg % NBUF) * CHUNKB, tid);
            }

            // consume chunk g from LDS: 6 k-steps x 4 WMMA
            const char* buf = smem + (size_t)(g % NBUF) * CHUNKB;
#pragma unroll
            for (int s = 0; s < CHK; ++s) {
                const int k = c * CHK + s;
                const char* lp = buf + (size_t)s * STEPB + (size_t)hi16 * 32;
                // B frag: lane L(<16): N=L, 16 contiguous K halves; lanes 16-31: K+16
                v16h b0 = *(const v16h*)(lp + (size_t)(0 * 16 + lo16) * ROWB);
                v16h b1 = *(const v16h*)(lp + (size_t)(1 * 16 + lo16) * ROWB);
                v16h b2 = *(const v16h*)(lp + (size_t)(2 * 16 + lo16) * ROWB);
                v16h b3 = *(const v16h*)(lp + (size_t)(3 * 16 + lo16) * ROWB);
                acc[0] = __builtin_amdgcn_wmma_f32_16x16x32_f16(false, afrag[k], false, b0, (short)0, acc[0], false, false);
                acc[1] = __builtin_amdgcn_wmma_f32_16x16x32_f16(false, afrag[k], false, b1, (short)0, acc[1], false, false);
                acc[2] = __builtin_amdgcn_wmma_f32_16x16x32_f16(false, afrag[k], false, b2, (short)0, acc[2], false, false);
                acc[3] = __builtin_amdgcn_wmma_f32_16x16x32_f16(false, afrag[k], false, b3, (short)0, acc[3], false, false);
            }
        }

        // D layout: VGPR v -> lanes 0-15: (M=v, N=lane); lanes 16-31: (M=v+8, N=lane-16)
#pragma unroll
        for (int ns = 0; ns < 4; ++ns) {
            unsigned r = rbase + (unsigned)ns * 16u + (unsigned)lo16;
            bool  ok = r < NPATCH;
            float sc = ok ? sbase[r] : 0.f;
#pragma unroll
            for (int v = 0; v < 8; ++v) {
                float val = acc[ns][v] * sc;
                if (ok && (val > bestVal[v] || (val == bestVal[v] && r < bestIdx[v]))) {
                    bestVal[v] = val; bestIdx[v] = r;
                }
            }
        }
    }

    // reduce across the 16 lanes of each half, then merge across r-chunks atomically
#pragma unroll
    for (int v = 0; v < 8; ++v) {
        float bv = bestVal[v]; unsigned bi = bestIdx[v];
#pragma unroll
        for (int m = 1; m < 16; m <<= 1) {
            float    ov = __shfl_xor(bv, m, 32);
            unsigned oi = __shfl_xor(bi, m, 32);
            if (ov > bv || (ov == bv && oi < bi)) { bv = ov; bi = oi; }
        }
        if (lo16 == 0 && bi != 0xFFFFFFFFu) {
            unsigned row = Mbase + (unsigned)v + (unsigned)hi16 * 8u;
            if (row < NPATCH) {
                unsigned u = __float_as_uint(bv);
                u = (u & 0x80000000u) ? ~u : (u | 0x80000000u);  // sortable float
                unsigned long long pk =
                    ((unsigned long long)u << 32) | (unsigned long long)(0xFFFFFFFFu - bi);
                atomicMax(&best[(size_t)n * NPAD + row], pk);     // ties -> smallest r wins
            }
        }
    }
}

// --------------------------------------------- decode argmax -> flow + shifts
__global__ void flow_out_kernel(const unsigned long long* __restrict__ best,
                                float* __restrict__ out)
{
    int t = blockIdx.x * blockDim.x + threadIdx.x;
    if (t >= NB * HW) return;
    int n = t / HW, q = t % HW, y = q / WW, x = q % WW;
#pragma unroll
    for (int s = 0; s < 9; ++s) {
        int i = s / 3, j = s % 3;
        int y0 = y - i, x0 = x - j;
        float fh = 0.f, fw = 0.f;
        if (y0 >= 0 && x0 >= 0 && y0 < HP && x0 < WP) {
            unsigned long long pk = best[(size_t)n * NPAD + (size_t)y0 * WP + x0];
            unsigned r = 0xFFFFFFFFu - (unsigned)(pk & 0xFFFFFFFFull);
            fw = (float)(int)(r % WP) - (float)x0;
            fh = (float)(int)(r / WP) - (float)y0;
        }
        out[(((size_t)n * 18 + 2 * s    ) * HH + y) * WW + x] = fh;  // even ch: flow_h
        out[(((size_t)n * 18 + 2 * s + 1) * HH + y) * WW + x] = fw;  // odd ch:  flow_w
    }
}

// ----------------------------------------------------------------- launcher
extern "C" void kernel_launch(void* const* d_in, const int* in_sizes, int n_in,
                              void* d_out, int out_size, void* d_ws, size_t ws_size,
                              hipStream_t stream) {
    const float* f1 = (const float*)d_in[0];
    const float* f2 = (const float*)d_in[1];
    float* out = (float*)d_out;

    char* ws = (char*)d_ws;
    size_t off = 0;
    auto take = [&](size_t bytes) { char* p = ws + off; off = (off + bytes + 255) & ~(size_t)255; return p; };
    float*              normF1 = (float*)take((size_t)NB * CCH * HW * 4);
    float*              normF2 = (float*)take((size_t)NB * CCH * HW * 4);
    _Float16*           Apat   = (_Float16*)take((size_t)NB * NPAD * KDIM * 2);
    _Float16*           Bpat   = (_Float16*)take((size_t)NB * NPAD * KDIM * 2);
    float*              scaleB = (float*)take((size_t)NB * NPAD * 4);
    unsigned long long* bestB  = (unsigned long long*)take((size_t)NB * NPAD * 8);
    (void)ws_size; (void)in_sizes; (void)n_in; (void)out_size;

    {   // init argmax buffer
        int cnt = NB * NPAD;
        zero_best_kernel<<<(cnt + 255) / 256, 256, 0, stream>>>(bestB, cnt);
    }
    {   // channel normalization
        int cnt = NB * HW;
        normalize_kernel<<<(cnt + 255) / 256, 256, 0, stream>>>(f1, normF1);
        normalize_kernel<<<(cnt + 255) / 256, 256, 0, stream>>>(f2, normF2);
    }
    {   // patch matrices (f16)
        int cnt = NB * NPAD * CCH;
        build_patches_kernel<<<(cnt + 255) / 256, 256, 0, stream>>>(normF1, Apat);
        build_patches_kernel<<<(cnt + 255) / 256, 256, 0, stream>>>(normF2, Bpat);
    }
    {   // per-ref-patch scales (f32)
        dim3 grid(NPATCH, NB);
        patch_scale_kernel<<<grid, 32, 0, stream>>>(normF2, scaleB);
    }
    {   // WMMA correlation + argmax (async-LDS pipelined)
        dim3 grid(MBLK, RSPLIT, NB);
        corr_argmax_kernel<<<grid, 512, SMEMB, stream>>>(Apat, Bpat, scaleB, bestB);
    }
    {   // flow decode + shifts + reorder
        int cnt = NB * HW;
        flow_out_kernel<<<(cnt + 255) / 256, 256, 0, stream>>>(bestB, out);
    }
}
